// QuantumMultiHeadAttention_89249420411142
// MI455X (gfx1250) — compile-verified, matrix operand order checked by reading, not run on hardware
//
#include <hip/hip_runtime.h>
#include <hip/hip_bf16.h>

// ---- problem constants ----
#define SQ   2048
#define DIM  1024
#define NH   16
#define HD   64
#define EPSQ 1e-10f

typedef __bf16 bf16;
typedef __attribute__((ext_vector_type(8)))  __bf16 v8bf;
typedef __attribute__((ext_vector_type(16))) __bf16 v16bf;
typedef __attribute__((ext_vector_type(8)))  float  v8f;
typedef __attribute__((ext_vector_type(4)))  unsigned int u32x4;
typedef __attribute__((ext_vector_type(8)))  int   i32x8;
typedef __attribute__((ext_vector_type(4)))  int   i32x4;

static __device__ inline v16bf mk16(v8bf lo, v8bf hi) {
  v16bf r;
#pragma unroll
  for (int i = 0; i < 8; ++i) { r[i] = lo[i]; r[8 + i] = hi[i]; }
  return r;
}

static __device__ inline float rmax16(float v) {
#pragma unroll
  for (int o = 8; o; o >>= 1) v = fmaxf(v, __shfl_xor(v, o, 16));
  return v;
}
static __device__ inline float rmin16(float v) {
#pragma unroll
  for (int o = 8; o; o >>= 1) v = fminf(v, __shfl_xor(v, o, 16));
  return v;
}
static __device__ inline float rsum16(float v) {
#pragma unroll
  for (int o = 8; o; o >>= 1) v += __shfl_xor(v, o, 16);
  return v;
}

// ---- TDM: 2D tile (tile_d0 elems of 2B contiguous, tile_d1 rows, row stride
// stride0 elems) from global -> LDS at byte offset lds_off. Issued per-wave. ----
#if __has_builtin(__builtin_amdgcn_tensor_load_to_lds)
#define QMHA_HAVE_TDM 1
static __device__ inline void tdm_load_2d(const bf16* gsrc, unsigned int lds_off,
                                          unsigned int tensor_d0, unsigned int tensor_d1,
                                          unsigned int tile_d0, unsigned int tile_d1,
                                          unsigned long long stride0_elems) {
  const unsigned long long ga = (unsigned long long)(uintptr_t)gsrc;
  u32x4 g0;
  g0[0] = 1u;                                   // count=1, user descriptor
  g0[1] = lds_off;                              // LDS byte address
  g0[2] = (unsigned int)ga;                     // global_addr[31:0]
  g0[3] = (unsigned int)(ga >> 32) | (2u << 30);// global_addr[56:32] | type=2
  i32x8 g1;
  g1[0] = (int)(1u << 16);                      // wg_mask=0, data_size=1 (2 bytes)
  g1[1] = (int)((tensor_d0 & 0xffffu) << 16);   // abar_addr=0 | tensor_dim0[15:0]
  g1[2] = (int)((tensor_d0 >> 16) | ((tensor_d1 & 0xffffu) << 16));
  g1[3] = (int)((tensor_d1 >> 16) | (tile_d0 << 16));
  g1[4] = (int)(tile_d1 & 0xffffu);             // tile_dim1 | tile_dim2=0
  g1[5] = (int)(unsigned int)stride0_elems;     // tensor_dim0_stride[31:0]
  g1[6] = (int)((unsigned int)(stride0_elems >> 32) & 0xffffu); // |dim1_stride=0
  g1[7] = 0;
  const i32x4 z4 = {0, 0, 0, 0};
#if defined(__clang_major__) && (__clang_major__ >= 23)
  const i32x8 z8 = {0, 0, 0, 0, 0, 0, 0, 0};
  __builtin_amdgcn_tensor_load_to_lds(g0, g1, z4, z4, z8, 0);
#else
  __builtin_amdgcn_tensor_load_to_lds(g0, g1, z4, z4, 0);
#endif
}
#else
#define QMHA_HAVE_TDM 0
#endif

// ---------------- f32 -> bf16 convert (straight) ----------------
__global__ __launch_bounds__(256) void qmha_cvt_bf16(const float* __restrict__ in,
                                                     bf16* __restrict__ out, int n) {
  for (int i = blockIdx.x * 256 + threadIdx.x; i < n; i += gridDim.x * 256)
    out[i] = (bf16)in[i];
}

// ---------------- f32 -> bf16 convert + transpose (square RxC) ----------------
// out[c*R + r] = (bf16) in[r*C + c];  64x64 LDS tiles, coalesced both sides.
__global__ __launch_bounds__(256) void qmha_cvt_bf16_t(const float* __restrict__ in,
                                                       bf16* __restrict__ out,
                                                       int R, int C) {
  __shared__ bf16 t[64][72];
  const int c0 = blockIdx.x * 64, r0 = blockIdx.y * 64;
  const int tx = threadIdx.x & 63, ty0 = threadIdx.x >> 6;
#pragma unroll
  for (int i = 0; i < 16; ++i) {
    const int ty = ty0 + i * 4;
    t[ty][tx] = (bf16)in[(size_t)(r0 + ty) * C + c0 + tx];
  }
  __syncthreads();
#pragma unroll
  for (int i = 0; i < 16; ++i) {
    const int ty = ty0 + i * 4;
    out[(size_t)(c0 + ty) * R + r0 + tx] = t[tx][ty];
  }
}

// ---------------- bf16 WMMA GEMM --------------------
// C[M,N] = A[M,K] * Bt[N,K]^T   (Bt is the PRE-TRANSPOSED right operand, [N][K] row-major)
// mode 0: C bf16 row-major; mode 1: C bf16 transposed ([N][M]); mode 2: C f32 row-major.
// Block tile 128x128, BK=32, 256 threads = 8 waves in 2x4 grid (each wave 64x32).
// Staging: double-buffered TDM (tensor_load_to_lds); DMA of tile t+1 overlaps WMMA on t.
__global__ __launch_bounds__(256) void qmha_gemm_bf16(const bf16* __restrict__ A,
                                                      const bf16* __restrict__ Bt,
                                                      void* __restrict__ C,
                                                      int M, int N, int K, int mode) {
  __shared__ bf16 As[2][128 * 32];    // [buf][m][k]
  __shared__ bf16 Bs[2][128 * 32];    // [buf][n][k]
  const int tid  = threadIdx.x;
  const int lane = tid & 31;
  const int wid  = tid >> 5;
  const int l15  = lane & 15;
  const int sel  = lane >> 4;
  const int waveM = wid >> 2;         // 0..1
  const int waveN = wid & 3;          // 0..3
  const int mBase = blockIdx.y * 128;
  const int nBase = blockIdx.x * 128;

  v8f acc[4][2];
#pragma unroll
  for (int i = 0; i < 4; ++i)
#pragma unroll
    for (int j = 0; j < 2; ++j) acc[i][j] = (v8f)0.0f;

#if QMHA_HAVE_TDM
  const unsigned int asOff0 = (unsigned int)(uintptr_t)(void*)&As[0][0];
  const unsigned int asOff1 = (unsigned int)(uintptr_t)(void*)&As[1][0];
  const unsigned int bsOff0 = (unsigned int)(uintptr_t)(void*)&Bs[0][0];
  const unsigned int bsOff1 = (unsigned int)(uintptr_t)(void*)&Bs[1][0];
  // prologue: DMA tile 0 into buffer 0
  if (wid == 0) {
    tdm_load_2d(&A [(size_t)mBase * K], asOff0, (unsigned)K, 128u, 32u, 128u,
                (unsigned long long)K);
    tdm_load_2d(&Bt[(size_t)nBase * K], bsOff0, (unsigned)K, 128u, 32u, 128u,
                (unsigned long long)K);
  }
#endif

  int cur = 0;
  for (int k0 = 0; k0 < K; k0 += 32) {
#if QMHA_HAVE_TDM
    if (wid == 0) {
      if (k0 + 32 < K) {
        // issue next tile into the other buffer, then wait only for current pair
        const unsigned int ao = cur ? asOff0 : asOff1;
        const unsigned int bo = cur ? bsOff0 : bsOff1;
        tdm_load_2d(&A [(size_t)mBase * K + k0 + 32], ao, (unsigned)K, 128u, 32u, 128u,
                    (unsigned long long)K);
        tdm_load_2d(&Bt[(size_t)nBase * K + k0 + 32], bo, (unsigned)K, 128u, 32u, 128u,
                    (unsigned long long)K);
        __builtin_amdgcn_s_wait_tensorcnt(2);
      } else {
        __builtin_amdgcn_s_wait_tensorcnt(0);
      }
    }
#else
    // fallback: 512 x 16B chunk copies (both tiles have identical layout)
    for (int c = tid; c < 512; c += 256) {
      const int row = c >> 2, sub = c & 3;
      *(uint4*)(&As[cur][row * 32 + sub * 8]) =
          *(const uint4*)(&A[(size_t)(mBase + row) * K + k0 + sub * 8]);
      *(uint4*)(&Bs[cur][row * 32 + sub * 8]) =
          *(const uint4*)(&Bt[(size_t)(nBase + row) * K + k0 + sub * 8]);
    }
#endif
    __syncthreads();

    const bf16* asp = &As[cur][0];
    const bf16* bsp = &Bs[cur][0];
    v16bf af[4], bfm[2];
#pragma unroll
    for (int mi = 0; mi < 4; ++mi) {
      const int r = waveM * 64 + mi * 16 + l15;
      af[mi] = mk16(*(const v8bf*)(&asp[r * 32 + sel * 8]),
                    *(const v8bf*)(&asp[r * 32 + 16 + sel * 8]));
    }
#pragma unroll
    for (int ni = 0; ni < 2; ++ni) {
      const int n = waveN * 32 + ni * 16 + l15;
      bfm[ni] = mk16(*(const v8bf*)(&bsp[n * 32 + sel * 8]),
                     *(const v8bf*)(&bsp[n * 32 + 16 + sel * 8]));
    }
#pragma unroll
    for (int mi = 0; mi < 4; ++mi)
#pragma unroll
      for (int ni = 0; ni < 2; ++ni)
        acc[mi][ni] = __builtin_amdgcn_wmma_f32_16x16x32_bf16(
            false, af[mi], false, bfm[ni], (short)0, acc[mi][ni], false, false);
    __syncthreads();
    cur ^= 1;
  }

  // write back
#pragma unroll
  for (int mi = 0; mi < 4; ++mi)
#pragma unroll
    for (int ni = 0; ni < 2; ++ni)
#pragma unroll
      for (int j = 0; j < 8; ++j) {
        const int row = mBase + waveM * 64 + mi * 16 + (sel ? j + 8 : j);
        const int col = nBase + waveN * 32 + ni * 16 + l15;
        const float v = acc[mi][ni][j];
        if (mode == 0)      ((bf16*)C)[(size_t)row * N + col] = (bf16)v;
        else if (mode == 1) ((bf16*)C)[(size_t)col * M + row] = (bf16)v;
        else                ((float*)C)[(size_t)row * N + col] = v;
      }
}

// ---------------- fused attention + quantum softmax ----------------
// grid: (SQ/16, NH). 256 threads = 8 waves. One (head, 16-query-row) strip per block.
// Dynamic LDS: sc[16][2048] f32 (128KB) | pb[16][2048] bf16 (64KB) | red[16][64] f32 (4KB)
__global__ __launch_bounds__(256) void qmha_attn(const bf16* __restrict__ Qb,
                                                 const bf16* __restrict__ Kb,
                                                 const bf16* __restrict__ Vt,   // [DIM][SQ]
                                                 bf16* __restrict__ ctxb) {
  extern __shared__ char smem[];
  float* sc = (float*)smem;                                    // 16*2048 f32
  bf16*  pb = (bf16*)(smem + 16 * 2048 * 4);                   // 16*2048 bf16
  float* red = (float*)(smem + 16 * 2048 * 4 + 16 * 2048 * 2); // 16*64 f32

  const int h   = blockIdx.y;
  const int qb0 = blockIdx.x * 16;
  const int tid = threadIdx.x;
  const int lane = tid & 31, wid = tid >> 5;
  const int l15 = lane & 15, sel = lane >> 4;

  // ---- phase A: scores = (Q K^T) * 1/sqrt(HD), strip [16][2048] into LDS ----
  const bf16* qptr = Qb + (size_t)(qb0 + l15) * DIM + h * HD;
  const v16bf a0 = mk16(*(const v8bf*)(qptr + sel * 8),
                        *(const v8bf*)(qptr + 16 + sel * 8));
  const v16bf a1 = mk16(*(const v8bf*)(qptr + 32 + sel * 8),
                        *(const v8bf*)(qptr + 48 + sel * 8));
  const int colBase = wid * 256;
#pragma unroll 4
  for (int nt = 0; nt < 16; ++nt) {
    const int n = colBase + nt * 16 + l15;
    const bf16* kptr = Kb + (size_t)n * DIM + h * HD;
    const v16bf b0 = mk16(*(const v8bf*)(kptr + sel * 8),
                          *(const v8bf*)(kptr + 16 + sel * 8));
    const v16bf b1 = mk16(*(const v8bf*)(kptr + 32 + sel * 8),
                          *(const v8bf*)(kptr + 48 + sel * 8));
    v8f c = (v8f)0.0f;
    c = __builtin_amdgcn_wmma_f32_16x16x32_bf16(false, a0, false, b0, (short)0, c, false, false);
    c = __builtin_amdgcn_wmma_f32_16x16x32_bf16(false, a1, false, b1, (short)0, c, false, false);
#pragma unroll
    for (int j = 0; j < 8; ++j)
      sc[(sel ? j + 8 : j) * 2048 + colBase + nt * 16 + l15] = c[j] * 0.125f;
  }
  __syncthreads();

  // ---- phase B: quantum softmax, one row per 16-thread group ----
  const int r = tid >> 4, cl = tid & 15;
  float* row = sc + r * 2048;

  float mx = -3.4e38f, mn = 3.4e38f;
  for (int i = cl; i < 2048; i += 16) {
    const float v = row[i];
    mx = fmaxf(mx, v); mn = fminf(mn, v);
  }
  mx = rmax16(mx); mn = rmin16(mn);
  const float smin = mn - mx;               // min of stabilized s (max is 0)
  const float rng  = -smin + EPSQ;

  const float c0 = 1.266065877752008f, c1 = 1.13031820798497f,
              c2 = 0.271495339534077f, c3 = 0.044336849848664f,
              c4 = 0.005474240442094f;

  float sT = 0.f, sP = 0.f, sC = 0.f, lT = 0.f, lP = 0.f, lC = 0.f;
  for (int i = cl; i < 2048; i += 16) {
    const float x = row[i] - mx;
    const float t5 = 1.f + x * (1.f + x * (0.5f + x * (0.16666667f +
                     x * (0.04166667f + x * 0.00833333333f))));
    const float x2 = x * x;
    const float p = fmaxf((1.f + x + 0.5f * x2) / ((1.f - x + 0.5f * x2) + EPSQ), 0.f);
    const float xn = 2.f * (x - smin) / rng - 1.f;
    const float T2 = 2.f * xn * xn - 1.f;
    const float T3 = 2.f * xn * T2 - xn;
    const float T4 = 2.f * xn * T3 - T2;
    const float ch = fmaxf(c0 + c1 * xn + c2 * T2 + c3 * T3 + c4 * T4, 0.f);
    sT += t5; sP += p; sC += ch;
    lT += __logf(t5 + EPSQ); lP += __logf(p + EPSQ); lC += __logf(ch + EPSQ);
  }
  sT = rsum16(sT); sP = rsum16(sP); sC = rsum16(sC);
  lT = rsum16(lT); lP = rsum16(lP); lC = rsum16(lC);

  const float u = 1.f / 2048.f;
  const float logu = -7.62461898f;          // log(1/2048)
  // mean log(a_norm) ~= mean log(a+eps) - log(sum+eps)
  const float mlT = lT * u - __logf(sT + EPSQ);
  const float mlP = lP * u - __logf(sP + EPSQ);
  const float mlC = lC * u - __logf(sC + EPSQ);
  const float wT = __expf(u * (mlT - logu));
  const float wP = __expf(u * (mlP - logu));
  const float wC = __expf(u * (mlC - logu));
  const float tot = wT + wP + wC + EPSQ;
  const float kT = wT / (tot * (sT + EPSQ));
  const float kP = wP / (tot * (sP + EPSQ));
  const float kC = wC / (tot * (sC + EPSQ));

  for (int i = cl; i < 2048; i += 16) {
    const float x = row[i] - mx;
    const float t5 = 1.f + x * (1.f + x * (0.5f + x * (0.16666667f +
                     x * (0.04166667f + x * 0.00833333333f))));
    const float x2 = x * x;
    const float p = fmaxf((1.f + x + 0.5f * x2) / ((1.f - x + 0.5f * x2) + EPSQ), 0.f);
    const float xn = 2.f * (x - smin) / rng - 1.f;
    const float T2 = 2.f * xn * xn - 1.f;
    const float T3 = 2.f * xn * T2 - xn;
    const float T4 = 2.f * xn * T3 - T2;
    const float ch = fmaxf(c0 + c1 * xn + c2 * T2 + c3 * T3 + c4 * T4, 0.f);
    pb[r * 2048 + i] = (bf16)(kT * t5 + kP * p + kC * ch);
  }
  __syncthreads();

  // ---- phase C: ctx = probs @ V  (per-wave k-slice, LDS f32 reduce) ----
  for (int i = tid; i < 16 * 64; i += 256) red[i] = 0.f;
  __syncthreads();

  v8f acc4[4];
#pragma unroll
  for (int i = 0; i < 4; ++i) acc4[i] = (v8f)0.0f;
  const int kB = wid * 256;
#pragma unroll 2
  for (int ks = 0; ks < 8; ++ks) {
    const int kk = kB + ks * 32;
    const bf16* pa = pb + (size_t)l15 * 2048 + kk;
    const v16bf a = mk16(*(const v8bf*)(pa + sel * 8),
                         *(const v8bf*)(pa + 16 + sel * 8));
#pragma unroll
    for (int nt = 0; nt < 4; ++nt) {
      const bf16* pv = Vt + (size_t)(h * HD + nt * 16 + l15) * SQ + kk;
      const v16bf b = mk16(*(const v8bf*)(pv + sel * 8),
                           *(const v8bf*)(pv + 16 + sel * 8));
      acc4[nt] = __builtin_amdgcn_wmma_f32_16x16x32_bf16(
          false, a, false, b, (short)0, acc4[nt], false, false);
    }
  }
#pragma unroll
  for (int nt = 0; nt < 4; ++nt)
#pragma unroll
    for (int j = 0; j < 8; ++j)
      atomicAdd(&red[(sel ? j + 8 : j) * 64 + nt * 16 + l15], acc4[nt][j]);
  __syncthreads();

  for (int i = tid; i < 16 * 64; i += 256) {
    const int rr = i >> 6, cc = i & 63;
    ctxb[(size_t)(qb0 + rr) * DIM + h * HD + cc] = (bf16)red[i];
  }
}

// ---------------- launch ----------------
extern "C" void kernel_launch(void* const* d_in, const int* in_sizes, int n_in,
                              void* d_out, int out_size, void* d_ws, size_t ws_size,
                              hipStream_t stream) {
  const float* x  = (const float*)d_in[0];
  const float* Wq = (const float*)d_in[1];
  const float* Wk = (const float*)d_in[2];
  const float* Wv = (const float*)d_in[3];
  const float* Wo = (const float*)d_in[4];
  float* out = (float*)d_out;

  char* w = (char*)d_ws;
  bf16* xb   = (bf16*)(w);                    // S*D bf16 : 4 MB
  bf16* wqt  = (bf16*)(w + (4u << 20));       // D*D bf16 transposed : 2 MB each
  bf16* wkt  = (bf16*)(w + (6u << 20));
  bf16* wvt  = (bf16*)(w + (8u << 20));
  bf16* wot  = (bf16*)(w + (10u << 20));
  bf16* Qb   = (bf16*)(w + (12u << 20));      // S*D bf16 : 4 MB
  bf16* Kb   = (bf16*)(w + (16u << 20));
  bf16* Vt   = (bf16*)(w + (20u << 20));      // D*S bf16 (transposed) : 4 MB
  bf16* ctxb = (bf16*)(w + (24u << 20));      // S*D bf16 : 4 MB

  qmha_cvt_bf16<<<2048, 256, 0, stream>>>(x, xb, SQ * DIM);
  dim3 tg(DIM / 64, DIM / 64);
  qmha_cvt_bf16_t<<<tg, 256, 0, stream>>>(Wq, wqt, DIM, DIM);
  qmha_cvt_bf16_t<<<tg, 256, 0, stream>>>(Wk, wkt, DIM, DIM);
  qmha_cvt_bf16_t<<<tg, 256, 0, stream>>>(Wv, wvt, DIM, DIM);
  qmha_cvt_bf16_t<<<tg, 256, 0, stream>>>(Wo, wot, DIM, DIM);

  dim3 gg(DIM / 128, SQ / 128);
  qmha_gemm_bf16<<<gg, 256, 0, stream>>>(xb, wqt, Qb, SQ, DIM, DIM, 0);
  qmha_gemm_bf16<<<gg, 256, 0, stream>>>(xb, wkt, Kb, SQ, DIM, DIM, 0);
  qmha_gemm_bf16<<<gg, 256, 0, stream>>>(xb, wvt, Vt, SQ, DIM, DIM, 1);

  const size_t attn_lds = 16 * 2048 * 4 + 16 * 2048 * 2 + 16 * 64 * 4; // 200704 B
  hipFuncSetAttribute((const void*)qmha_attn,
                      hipFuncAttributeMaxDynamicSharedMemorySize, (int)attn_lds);
  qmha_attn<<<dim3(SQ / 16, NH), 256, attn_lds, stream>>>(Qb, Kb, Vt, ctxb);

  qmha_gemm_bf16<<<gg, 256, 0, stream>>>(ctxb, wot, (void*)out, SQ, DIM, DIM, 2);
}
